// DecoderRNN_29798483099660
// MI455X (gfx1250) — matrix operand order repaired and checked by reading, not compiled, not run on hardware
//
#include <hip/hip_runtime.h>
#include <hip/hip_bf16.h>
#include <cstdint>
#include <cstddef>

// ---------------------------------------------------------------------------
// GRU decoder (teacher forced), 160 steps, B=2048, H=512, V=128, EMB=64.
//
// Strategy (MI455X / gfx1250, wave32, WMMA):
//  * One persistent kernel: 128 workgroups x 256 threads (8 waves).  Each WG
//    owns a 16-row batch tile and iterates all 160 timesteps locally -- the
//    recurrence is per-batch-row, so no inter-WG sync is ever needed.
//  * gi-GEMM eliminated: G[v][3H] = relu(emb[v]) @ W_ih^T + b_ih precomputed
//    once (VOCAB=128 -> 0.77MB table, L2 resident); per-step gi is a gather.
//  * gh and logits GEMMs use v_wmma_f32_16x16x32_bf16 (f32 accum).
//    h kept in f32 *registers* (each lane permanently owns its 32 elements),
//    mirrored as bf16 in double-buffered LDS for the WMMA A-operand.
//  * W_hh / W_out converted to bf16 once; B-fragments are K-contiguous
//    32-byte loads straight from L2 (no transpose needed: gh = h @ W_hh^T
//    means B[k][n] = W_hh[n][k], row-major rows are the K axis).
// ---------------------------------------------------------------------------

#define T_STEPS 160
#define HID     512
#define EMBD    64
#define VOC     128
#define BATCH   2048
#define BTILE   16
#define NBLK    (BATCH / BTILE)      // 128 workgroups
#define THREADS 256                  // 8 waves (wave32)

typedef __attribute__((ext_vector_type(16))) __bf16 v16bf;
typedef __attribute__((ext_vector_type(8)))  __bf16 v8bf;
typedef __attribute__((ext_vector_type(8)))  float  v8f;

// workspace layout (bytes); needs ~2.4 MB
#define G_ELEMS    (VOC * 3 * HID)          // 196608 f32
#define WHH_ELEMS  (3 * HID * HID)          // 786432 bf16
#define WOUT_ELEMS (VOC * HID)              // 65536  bf16
#define WS_G       0
#define WS_WHH     (G_ELEMS * 4)
#define WS_WOUT    (WS_WHH + WHH_ELEMS * 2)

// ---------------------------------------------------------------------------
// One-time prep: G table + bf16 weight conversion.
//  blocks [0,768)      : G[v][col] = b_ih[col] + sum_e relu(emb[v][e])*W_ih[col][e]
//  blocks [768,3840)   : W_hh  f32 -> bf16
//  blocks [3840,4096)  : W_out f32 -> bf16
// ---------------------------------------------------------------------------
__global__ __launch_bounds__(THREADS) void gru_prep_kernel(
    const float* __restrict__ emb,  const float* __restrict__ W_ih,
    const float* __restrict__ b_ih, const float* __restrict__ W_hh,
    const float* __restrict__ W_out,
    float* __restrict__ G, __bf16* __restrict__ WhhBf, __bf16* __restrict__ WoutBf)
{
    const int b = blockIdx.x, tid = threadIdx.x;
    if (b < 768) {
        const int idx = b * THREADS + tid;          // [0, 196608)
        const int v   = idx / (3 * HID);
        const int col = idx % (3 * HID);
        float acc = b_ih[col];
        const float* e = emb + (size_t)v * EMBD;
        const float* w = W_ih + (size_t)col * EMBD;
        #pragma unroll 8
        for (int k = 0; k < EMBD; ++k) {
            float x = e[k];
            x = x > 0.f ? x : 0.f;                  // relu(embedding)
            acc = fmaf(x, w[k], acc);
        }
        G[idx] = acc;
    } else if (b < 3840) {
        const int i = (b - 768) * THREADS + tid;    // [0, 786432)
        WhhBf[i] = (__bf16)W_hh[i];
    } else {
        const int i = (b - 3840) * THREADS + tid;   // [0, 65536)
        WoutBf[i] = (__bf16)W_out[i];
    }
}

// Build a 16x32 bf16 A-fragment (K-contiguous, matching B):
//  lane L<16  : row M=L,    K = {k0..k0+15}
//  lane L>=16 : row M=L-16, K = {k0+16..k0+31}
// caller passes rowbase = &h[row][k0 + (lane>=16 ? 16 : 0)]
__device__ __forceinline__ v16bf load_a_frag(const __bf16* rowbase) {
    v8bf lo = *reinterpret_cast<const v8bf*>(rowbase);
    v8bf hi = *reinterpret_cast<const v8bf*>(rowbase + 8);
    return __builtin_shufflevector(lo, hi, 0,1,2,3,4,5,6,7,8,9,10,11,12,13,14,15);
}

__global__ __launch_bounds__(THREADS) void gru_decoder_kernel(
    const float*  __restrict__ h0,      // encoder_hidden [1,B,H]
    const int*    __restrict__ target,  // [B, T]
    const float*  __restrict__ G,       // [V, 3H]  (bias-folded gi table)
    const __bf16* __restrict__ WhhBf,   // [3H, H]
    const __bf16* __restrict__ WoutBf,  // [V, H]
    const float*  __restrict__ b_hh,    // [3H]
    const float*  __restrict__ b_out,   // [V]
    float*        __restrict__ out)     // [T,B,V] log_probs ++ [B,H] h_last
{
    __shared__ __align__(32) __bf16 hbf[2][BTILE][HID];   // 32 KB double-buffered A-operand
    __shared__ float lgs[BTILE][VOC];                     // 8 KB logits staging
    __shared__ int   tokLDS[BTILE];
    __shared__ float rowMax[BTILE], rowLS[BTILE];

    const int tid  = threadIdx.x;
    const int wave = tid >> 5;
    const int lane = tid & 31;
    const int n    = lane & 15;     // WMMA N index (and A-row index)
    const int kh   = lane >> 4;     // K-half selector for fragments
    const int bb   = blockIdx.x * BTILE;

    // ---- h state: f32 in registers (lane owns (rows kh*8..kh*8+7) x col) ----
    v8f hreg[4];
    #pragma unroll
    for (int q = 0; q < 4; ++q) {
        const int col = (wave + 8 * q) * 16 + n;
        #pragma unroll
        for (int i = 0; i < 8; ++i)
            hreg[q][i] = h0[(size_t)(bb + kh * 8 + i) * HID + col];
    }
    for (int e = tid; e < BTILE * HID; e += THREADS) {
        const int m = e >> 9, c = e & (HID - 1);
        hbf[0][m][c] = (__bf16)h0[(size_t)(bb + m) * HID + c];
    }
    __syncthreads();

    const size_t OUT_H = (size_t)T_STEPS * BATCH * VOC;
    int cur = 0;

    for (int t = 0; t < T_STEPS; ++t) {
        const int nxt = cur ^ 1;
        if (tid < BTILE)
            tokLDS[tid] = (t == 0) ? 0 : target[(size_t)(bb + tid) * T_STEPS + (t - 1)];
        __syncthreads();  // tokens ready; prior readers of hbf[nxt] done

        // ---- gh = h @ W_hh^T ; gates ; h update.  4 hidden-col tiles/wave ----
        #pragma unroll
        for (int q = 0; q < 4; ++q) {
            const int col = (wave + 8 * q) * 16 + n;         // hidden unit j in [0,512)
            const __bf16* Br = WhhBf + (size_t)col * HID + kh * 16;            // r rows
            const __bf16* Bz = Br + (size_t)HID * HID;                         // z rows
            const __bf16* Bn = Bz + (size_t)HID * HID;                         // n rows
            const __bf16* Arow = &hbf[cur][n][kh * 16];

            v8f accR = {0.f,0.f,0.f,0.f,0.f,0.f,0.f,0.f};
            v8f accZ = accR, accN = accR;
            #pragma unroll 4
            for (int k0 = 0; k0 < HID; k0 += 32) {
                v16bf a  = load_a_frag(Arow + k0);
                v16bf br = *reinterpret_cast<const v16bf*>(Br + k0);
                v16bf bz = *reinterpret_cast<const v16bf*>(Bz + k0);
                v16bf bn = *reinterpret_cast<const v16bf*>(Bn + k0);
                accR = __builtin_amdgcn_wmma_f32_16x16x32_bf16(false, a, false, br, (short)0, accR, false, false);
                accZ = __builtin_amdgcn_wmma_f32_16x16x32_bf16(false, a, false, bz, (short)0, accZ, false, false);
                accN = __builtin_amdgcn_wmma_f32_16x16x32_bf16(false, a, false, bn, (short)0, accN, false, false);
            }
            const float bhr = b_hh[col];
            const float bhz = b_hh[HID + col];
            const float bhn = b_hh[2 * HID + col];
            #pragma unroll
            for (int i = 0; i < 8; ++i) {
                const int m = kh * 8 + i;                    // D-matrix row = batch row
                const float* g = G + (size_t)tokLDS[m] * (3 * HID) + col;
                const float r = 1.f / (1.f + __expf(-(g[0]        + accR[i] + bhr)));
                const float z = 1.f / (1.f + __expf(-(g[HID]      + accZ[i] + bhz)));
                const float nn = tanhf(      g[2 * HID]  + r * (accN[i] + bhn));
                const float hnew = (1.f - z) * nn + z * hreg[q][i];
                hreg[q][i] = hnew;
                hbf[nxt][m][col] = (__bf16)hnew;
            }
        }
        __syncthreads();  // h_new (bf16) complete

        // ---- logits = h_new @ W_out^T + b_out : one 16x16 vocab tile/wave ----
        {
            const int vcol = wave * 16 + n;
            const __bf16* Bv   = WoutBf + (size_t)vcol * HID + kh * 16;
            const __bf16* Arow = &hbf[nxt][n][kh * 16];
            v8f acc = {0.f,0.f,0.f,0.f,0.f,0.f,0.f,0.f};
            #pragma unroll 4
            for (int k0 = 0; k0 < HID; k0 += 32) {
                v16bf a = load_a_frag(Arow + k0);
                v16bf b = *reinterpret_cast<const v16bf*>(Bv + k0);
                acc = __builtin_amdgcn_wmma_f32_16x16x32_bf16(false, a, false, b, (short)0, acc, false, false);
            }
            const float bo = b_out[vcol];
            #pragma unroll
            for (int i = 0; i < 8; ++i)
                lgs[kh * 8 + i][vcol] = acc[i] + bo;
        }
        __syncthreads();  // logits staged

        // ---- log-softmax stats (one lane per batch row) ----
        if (tid < BTILE) {
            const float* row = lgs[tid];
            float mx = row[0];
            #pragma unroll 4
            for (int v = 1; v < VOC; ++v) mx = fmaxf(mx, row[v]);
            float s = 0.f;
            #pragma unroll 4
            for (int v = 0; v < VOC; ++v) s += __expf(row[v] - mx);
            rowMax[tid] = mx;
            rowLS[tid]  = __logf(s);
        }
        __syncthreads();  // stats ready

        // ---- write log_probs: 16x128 tile, 8 contiguous floats per thread ----
        {
            float* dst = out + ((size_t)t * BATCH + bb) * VOC;
            const int e0 = tid * 8;
            const int r  = e0 >> 7;
            const int v0 = e0 & (VOC - 1);
            const float sub = rowMax[r] + rowLS[r];
            #pragma unroll
            for (int c = 0; c < 8; ++c)
                dst[(size_t)r * VOC + v0 + c] = lgs[r][v0 + c] - sub;
        }
        cur = nxt;
    }

    // ---- final hidden state [1,B,H] appended after log_probs ----
    #pragma unroll
    for (int q = 0; q < 4; ++q) {
        const int col = (wave + 8 * q) * 16 + n;
        #pragma unroll
        for (int i = 0; i < 8; ++i)
            out[OUT_H + (size_t)(bb + kh * 8 + i) * HID + col] = hreg[q][i];
    }
}

extern "C" void kernel_launch(void* const* d_in, const int* in_sizes, int n_in,
                              void* d_out, int out_size, void* d_ws, size_t ws_size,
                              hipStream_t stream) {
    (void)in_sizes; (void)n_in; (void)out_size; (void)ws_size;
    // setup_inputs() order:
    // 0 encoder_outputs (unused by reference math), 1 encoder_hidden,
    // 2 target_tensor, 3 embedding, 4 W_ih, 5 W_hh, 6 b_ih, 7 b_hh, 8 W_out, 9 b_out
    const float* enc_hid = (const float*)d_in[1];
    const int*   target  = (const int*)d_in[2];
    const float* emb     = (const float*)d_in[3];
    const float* W_ih    = (const float*)d_in[4];
    const float* W_hh    = (const float*)d_in[5];
    const float* b_ih    = (const float*)d_in[6];
    const float* b_hh    = (const float*)d_in[7];
    const float* W_out   = (const float*)d_in[8];
    const float* b_out   = (const float*)d_in[9];

    char* ws = (char*)d_ws;
    float*  G      = (float*)(ws + WS_G);
    __bf16* WhhBf  = (__bf16*)(ws + WS_WHH);
    __bf16* WoutBf = (__bf16*)(ws + WS_WOUT);

    gru_prep_kernel<<<4096, THREADS, 0, stream>>>(emb, W_ih, b_ih, W_hh, W_out,
                                                  G, WhhBf, WoutBf);
    gru_decoder_kernel<<<NBLK, THREADS, 0, stream>>>(enc_hid, target, G, WhhBf, WoutBf,
                                                     b_hh, b_out, (float*)d_out);
}
